// VisionEncoder_46746424050243
// MI455X (gfx1250) — compile-verified, matrix operand order checked by reading, not run on hardware
//
#include <hip/hip_runtime.h>
#include <hip/hip_bf16.h>

// ---------------- constants (reference shapes) ----------------
static constexpr int kD    = 768;
static constexpr int kH    = 12;
static constexpr int kNL   = 12;
static constexpr int kHD   = 64;     // head dim
static constexpr int kB    = 16;
static constexpr int kIMG  = 224;
static constexpr int kP    = 16;
static constexpr int kNP   = 196;    // patches
static constexpr int kL0   = 197;    // tokens after cls
static constexpr int kOUT  = 512;
static constexpr int kLCAP = 208;    // padded sequence capacity (mult of 16)
static constexpr int kLC   = 224;    // padded attention K length (mult of 32)
static constexpr int kSZ   = 3;      // context window
static constexpr float kEPS = 1e-5f;

typedef __attribute__((ext_vector_type(16))) __bf16 v16bf;
typedef __attribute__((ext_vector_type(8)))  float  v8f;
typedef __attribute__((ext_vector_type(4)))  unsigned int u32x4;
typedef __attribute__((ext_vector_type(8)))  int i32x8;
typedef __attribute__((ext_vector_type(4)))  int i32x4;

struct __align__(16) U128 { unsigned int a, b, c, d; };
union FragBF { v16bf v; U128 q[2]; unsigned short s[16]; };

__device__ __forceinline__ int imin(int a, int b) { return a < b ? a : b; }
__device__ __forceinline__ int imax(int a, int b) { return a > b ? a : b; }

__device__ __forceinline__ unsigned short f2bf(float f) {
  unsigned u = __float_as_uint(f);
  u = u + 0x7FFFu + ((u >> 16) & 1u);   // round-to-nearest-even
  return (unsigned short)(u >> 16);
}
__device__ __forceinline__ float bf2f(unsigned short h) {
  return __uint_as_float(((unsigned)h) << 16);
}

// ---------------- Tensor Data Mover availability ----------------
#if defined(__HIP_DEVICE_COMPILE__) && defined(__has_builtin)
#if __has_builtin(__builtin_amdgcn_tensor_load_to_lds) && __has_builtin(__builtin_amdgcn_s_wait_tensorcnt)
#define HAS_TDM 1
#endif
#endif

#ifdef HAS_TDM
// Issue one TDM 2-D tile load: rows x kcElems bf16 elements, row stride
// rowStrideElems, from gsrc into LDS byte offset ldsByteOff (compact tile).
// Descriptor packing per CDNA5 ISA 8.3/8.4 (group0: count/lds/global/type,
// group1: data_size=2B, tensor dims == tile dims, dim0 stride).
__device__ __forceinline__ void tdm_load_tile(unsigned ldsByteOff,
    const unsigned short* gsrc, int kcElems, int rows, long long rowStrideElems) {
  unsigned long long ga = (unsigned long long)gsrc;
  u32x4 g0;
  g0[0] = 1u;                                            // count=1, user D#
  g0[1] = ldsByteOff;                                    // lds_addr (bytes)
  g0[2] = (unsigned)(ga & 0xFFFFFFFFull);                // global_addr lo
  g0[3] = (unsigned)((ga >> 32) & 0x1FFFFFFull) | (2u << 30); // addr hi | type=2
  i32x8 g1;
  g1[0] = 1 << 16;                                       // data_size=1 (2 bytes)
  g1[1] = (int)((unsigned)kcElems << 16);                // tensor_dim0 lo16
  g1[2] = (int)((unsigned)rows << 16);                   // dim0 hi16=0 | tensor_dim1 lo16
  g1[3] = (int)((unsigned)kcElems << 16);                // dim1 hi16=0 | tile_dim0
  g1[4] = rows;                                          // tile_dim1 | tile_dim2=0
  unsigned long long st = (unsigned long long)rowStrideElems;
  g1[5] = (int)(st & 0xFFFFFFFFull);                     // dim0_stride lo32
  g1[6] = (int)((st >> 32) & 0xFFFFull);                 // dim0_stride hi16
  g1[7] = 0;
  i32x4 z4; z4[0] = 0; z4[1] = 0; z4[2] = 0; z4[3] = 0;
#if __has_include(<hip/amd_detail/amd_gfx1250_TDM.h>)
  i32x8 z8; z8[0]=0; z8[1]=0; z8[2]=0; z8[3]=0; z8[4]=0; z8[5]=0; z8[6]=0; z8[7]=0;
  __builtin_amdgcn_tensor_load_to_lds(g0, g1, z4, z4, z8, 0);   // clang-23 / therock
#else
  __builtin_amdgcn_tensor_load_to_lds(g0, g1, z4, z4, 0);       // ROCm 7.2 clang-22
#endif
}
#endif

// ---------------- generic WMMA GEMM ----------------
// C[M,N](f32 and/or bf16) = epilogue( scale * A[M,K](bf16) @ W^T + bias )
// W row n is K-contiguous at W + n*ldwn (ALL operands pre-arranged K-major).
// Block: 128 thr / 4 waves. Block tile 32(M) x 256(N); wave tile 32x64.
// A tile (32 x KC<=128) staged into LDS by the Tensor Data Mover, double
// buffered: issue tile i+1, s_wait_tensorcnt 1 -> tile i ready, barrier, MACs.
// Batched via blockIdx.z = zb*zh + zhh with independent strides.
// op: 0=none 1=+bias 2=+bias+resid 3=+bias,QGELU 4=+bias,ReLU
// NOTE: A tiles over-read up to 16 rows past M -> A buffers carry tail pads.
struct GemmP {
  const unsigned short* A;
  const unsigned short* Wt;
  const float* bias;
  const float* resid;
  float* C;
  unsigned short* Cb;
  long long ldwn;
  long long sAb, sAh, sWb, sWh, sCb, sCh;
  int M, N, K, lda, ldc, ldres, zh, op;
  float scale;
};

__global__ __launch_bounds__(128) void wmma_gemm_bf16(GemmP p) {
  __shared__ unsigned short lAs[2][32 * 128];   // 2 x 8 KB A tiles (double buffer)

  const int wave = threadIdx.x >> 5;
  const int lane = threadIdx.x & 31;
  const int half = lane >> 4;
  const int l16  = lane & 15;
  const int zb   = (int)blockIdx.z / p.zh;
  const int zhh  = (int)blockIdx.z % p.zh;

  const unsigned short* A = p.A + (long long)zb * p.sAb + (long long)zhh * p.sAh;
  const unsigned short* W = p.Wt + (long long)zb * p.sWb + (long long)zhh * p.sWh;
  const long long coff = (long long)zb * p.sCb + (long long)zhh * p.sCh;

  const int tileM  = blockIdx.y * 32;
  const int tileN0 = blockIdx.x * 256 + wave * 64;
  const unsigned short* atile = A + (long long)tileM * p.lda;

  int ncol[4];
#pragma unroll
  for (int t = 0; t < 4; t++) ncol[t] = imin(tileN0 + t * 16 + l16, p.N - 1);

  v8f acc0[4], acc1[4];
#pragma unroll
  for (int t = 0; t < 4; t++)
#pragma unroll
    for (int r = 0; r < 8; r++) { acc0[t][r] = 0.f; acc1[t][r] = 0.f; }

#ifdef HAS_TDM
  const unsigned ldsBase = (unsigned)(unsigned long long)(void*)&lAs[0][0];
  if (wave == 0)  // prime the pipeline with tile 0
    tdm_load_tile(ldsBase, atile, imin(128, p.K), 32, p.lda);
#endif

  int stage = 0;
  for (int kc = 0; kc < p.K; kc += 128, stage ^= 1) {
    const int KC = imin(128, p.K - kc);

#ifdef HAS_TDM
    if (wave == 0) {
      if (kc + 128 < p.K) {   // issue next tile, then wait for current (<=1 in flight)
        tdm_load_tile(ldsBase + (unsigned)((stage ^ 1) * 32 * 128 * 2),
                      atile + kc + 128, imin(128, p.K - (kc + 128)), 32, p.lda);
        __builtin_amdgcn_s_wait_tensorcnt(1);
      } else {
        __builtin_amdgcn_s_wait_tensorcnt(0);
      }
    }
#else
    for (int i = threadIdx.x; i < (32 * KC) / 8; i += 128) {
      const int r = (i * 8) / KC, c = (i * 8) % KC;
      *(U128*)(&lAs[stage][0] + r * KC + c) =
          *(const U128*)(atile + (long long)r * p.lda + kc + c);
    }
#endif
    __syncthreads();

    const unsigned short* lA = &lAs[stage][0];
    for (int k0 = 0; k0 < KC; k0 += 32) {
      FragBF a0, a1;
      const unsigned short* ls = lA + k0 + half * 8;
      a0.q[0] = *(const U128*)(ls + l16 * KC);
      a0.q[1] = *(const U128*)(ls + l16 * KC + 16);
      a1.q[0] = *(const U128*)(ls + (16 + l16) * KC);
      a1.q[1] = *(const U128*)(ls + (16 + l16) * KC + 16);

#pragma unroll
      for (int t = 0; t < 4; t++) {
        FragBF b;
        const unsigned short* wr = W + (long long)ncol[t] * p.ldwn;
        b.q[0] = *(const U128*)(wr + kc + k0 + half * 8);
        b.q[1] = *(const U128*)(wr + kc + k0 + 16 + half * 8);
        if (t == 0) __builtin_prefetch(wr + kc + k0 + 32, 0, 1);
        acc0[t] = __builtin_amdgcn_wmma_f32_16x16x32_bf16(
            false, a0.v, false, b.v, (short)0, acc0[t], false, false);
        acc1[t] = __builtin_amdgcn_wmma_f32_16x16x32_bf16(
            false, a1.v, false, b.v, (short)0, acc1[t], false, false);
      }
    }
    __syncthreads();
  }

  // epilogue: C/D layout -> VGPR r holds row base + r + 8*half, col = tileN0 + t*16 + l16
#pragma unroll
  for (int m = 0; m < 2; m++) {
    const int rowBase = tileM + m * 16;
#pragma unroll
    for (int t = 0; t < 4; t++) {
      const int col = tileN0 + t * 16 + l16;
      if (col >= p.N) continue;
      const float bia = (p.op >= 1 && p.bias) ? p.bias[col] : 0.f;
#pragma unroll
      for (int r = 0; r < 8; r++) {
        const int row = rowBase + r + half * 8;
        if (row >= p.M) continue;
        float v = (m ? acc1[t][r] : acc0[t][r]) * p.scale + bia;
        if (p.op == 2)      v += p.resid[(long long)row * p.ldres + col];
        else if (p.op == 3) v = v / (1.f + __expf(-1.702f * v));   // QuickGELU
        else if (p.op == 4) v = v > 0.f ? v : 0.f;                  // ReLU
        const long long idx = coff + (long long)row * p.ldc + col;
        if (p.C)  p.C[idx]  = v;
        if (p.Cb) p.Cb[idx] = f2bf(v);
      }
    }
  }
}

// ---------------- LayerNorm (one block per row) ----------------
__global__ __launch_bounds__(128) void ln_kernel(
    const float* in, long long istride, const float* g, const float* bta,
    float* outf, long long ostride, unsigned short* outb, long long obstride, int n) {
  __shared__ float red[128];
  const float* row = in + (long long)blockIdx.x * istride;
  float s = 0.f;
  for (int i = threadIdx.x; i < n; i += 128) s += row[i];
  red[threadIdx.x] = s; __syncthreads();
  if (threadIdx.x == 0) { float t = 0; for (int i = 0; i < 128; i++) t += red[i]; red[0] = t / (float)n; }
  __syncthreads();
  const float mean = red[0];
  __syncthreads();
  float vv = 0.f;
  for (int i = threadIdx.x; i < n; i += 128) { float d2 = row[i] - mean; vv += d2 * d2; }
  red[threadIdx.x] = vv; __syncthreads();
  if (threadIdx.x == 0) { float t = 0; for (int i = 0; i < 128; i++) t += red[i]; red[0] = rsqrtf(t / (float)n + kEPS); }
  __syncthreads();
  const float rstd = red[0];
  for (int i = threadIdx.x; i < n; i += 128) {
    float val = (row[i] - mean) * rstd * g[i] + bta[i];
    if (outf) outf[(long long)blockIdx.x * ostride + i] = val;
    if (outb) outb[(long long)blockIdx.x * obstride + i] = f2bf(val);
  }
}

// ---------------- softmax row (f32 scores -> bf16 probs, zero-padded to Lc) ----------------
__global__ __launch_bounds__(128) void softmax_kernel(
    const float* sc, unsigned short* pb, int L, int Lc) {
  __shared__ float red[128];
  const long long base = (long long)blockIdx.x * Lc;
  float m = -1e30f;
  for (int i = threadIdx.x; i < L; i += 128) m = fmaxf(m, sc[base + i]);
  red[threadIdx.x] = m; __syncthreads();
  if (threadIdx.x == 0) { float t = -1e30f; for (int i = 0; i < 128; i++) t = fmaxf(t, red[i]); red[0] = t; }
  __syncthreads();
  m = red[0];
  __syncthreads();
  float s = 0.f;
  for (int i = threadIdx.x; i < L; i += 128) s += __expf(sc[base + i] - m);
  red[threadIdx.x] = s; __syncthreads();
  if (threadIdx.x == 0) { float t = 0; for (int i = 0; i < 128; i++) t += red[i]; red[0] = 1.f / t; }
  __syncthreads();
  const float inv = red[0];
  for (int i = threadIdx.x; i < Lc; i += 128) {
    float v = (i < L) ? __expf(sc[base + i] - m) * inv : 0.f;
    pb[base + i] = f2bf(v);
  }
}

// ---------------- small glue kernels ----------------
__global__ void cvt_bf16_kernel(const float* s, unsigned short* d, long long n) {
  long long i = (long long)blockIdx.x * 256 + threadIdx.x;
  if (i < n) d[i] = f2bf(s[i]);
}

// vt[b,h,d,k] = v[b,k,h,d]  (makes attn*V B-operand K-contiguous)
__global__ void transpose_v_kernel(const unsigned short* qkvb, unsigned short* vt) {
  long long i = (long long)blockIdx.x * 256 + threadIdx.x;
  const long long total = (long long)kB * kH * kHD * kLC;
  if (i >= total) return;
  int k = (int)(i % kLC); long long t = i / kLC;
  int d = (int)(t % kHD); long long t2 = t / kHD;
  int h = (int)(t2 % kH); int b = (int)(t2 / kH);
  vt[i] = qkvb[((long long)b * kLCAP + k) * 3 * kD + 2 * kD + (long long)h * kHD + d];
}

// projT[n,k] = proj[k,n]  (final projection B-operand K-contiguous)
__global__ void transpose_proj_kernel(const float* proj, unsigned short* pt) {
  long long i = (long long)blockIdx.x * 256 + threadIdx.x;
  const long long total = (long long)kOUT * kD;
  if (i >= total) return;
  int k = (int)(i % kD); int n = (int)(i / kD);
  pt[i] = f2bf(proj[(long long)k * kOUT + n]);
}

__global__ void im2col_kernel(const float* video, unsigned short* col) {
  long long i = (long long)blockIdx.x * 256 + threadIdx.x;
  const long long total = (long long)kB * kNP * 768;
  if (i >= total) return;
  int k = (int)(i % 768); long long t = i / 768;
  int pp = (int)(t % kNP); int b = (int)(t / kNP);
  int c = k >> 8, r = (k >> 4) & 15, cc = k & 15;
  int py = (pp / 14) * kP + r, px = (pp % 14) * kP + cc;
  col[i] = f2bf(video[(((long long)b * 3 + c) * kIMG + py) * kIMG + px]);
}

__global__ void assemble_kernel(const float* patch, const float* cls,
                                const float* pos, float* x) {
  long long i = (long long)blockIdx.x * 256 + threadIdx.x;
  const long long total = (long long)kB * kLCAP * kD;
  if (i >= total) return;
  int d = (int)(i % kD); long long t = i / kD;
  int l = (int)(t % kLCAP); int b = (int)(t / kLCAP);
  float v;
  if (l == 0)            v = cls[d] + pos[d];
  else if (l <= kNP)     v = patch[((long long)b * kNP + (l - 1)) * kD + d] + pos[(long long)l * kD + d];
  else                   v = 0.f;   // pad rows kept finite
  x[i] = v;
}

__global__ void copy_prompt_kernel(const float* prompt, float* x) {
  int i = blockIdx.x * 256 + threadIdx.x;
  if (i >= kB * kD) return;
  int d = i % kD, b = i / kD;
  x[((long long)b * kLCAP + kL0) * kD + d] = prompt[(long long)b * kD + d];
}

__global__ __launch_bounds__(256) void attn_reduce_argmax(
    const unsigned short* probs, float* aout, int* idx) {
  __shared__ float sv[256];
  __shared__ int   si[256];
  const int b = blockIdx.x, j = threadIdx.x;
  float a = -1e30f;
  if (j < kNP) {
    float s = 0.f;
    for (int h = 0; h < kH; h++)
      s += bf2f(probs[(((long long)(b * kH + h) * kLCAP) + kL0) * kLC + 1 + j]);
    aout[(long long)b * kNP + j] = s;
    a = s;
  }
  sv[j] = a; si[j] = j; __syncthreads();
  for (int st = 128; st > 0; st >>= 1) {
    if (j < st) {
      if (sv[j + st] > sv[j] || (sv[j + st] == sv[j] && si[j + st] < si[j])) {
        sv[j] = sv[j + st]; si[j] = si[j + st];
      }
    }
    __syncthreads();
  }
  if (j == 0) idx[b] = si[0];
}

__global__ void build_ctx_kernel(const float* x, const int* idx, const float* pos_ctx,
                                 float* t, unsigned short* tb, int li) {
  long long i = (long long)blockIdx.x * 256 + threadIdx.x;
  const long long total = (long long)kSZ * kB * kD;
  if (i >= total) return;
  int d = (int)(i % kD); int r = (int)(i / kD);
  int b = r % kB, j = r / kB;
  int bs = imin(imax(b + j - (kSZ / 2), 0), kB - 1);       // edge pad
  float v = x[((long long)bs * kLCAP + 1 + idx[bs]) * kD + d]
          + pos_ctx[(((long long)(li - 1)) * kSZ + j) * kD + d];
  t[i] = v; tb[i] = f2bf(v);
}

__global__ void scatter_ctx_kernel(const float* t2, float* x) {
  long long i = (long long)blockIdx.x * 256 + threadIdx.x;
  const long long total = (long long)kSZ * kB * kD;
  if (i >= total) return;
  int d = (int)(i % kD); int r = (int)(i / kD);
  int b = r % kB, j = r / kB;
  x[((long long)b * kLCAP + (kL0 + 1) + j) * kD + d] = t2[i];
}

__global__ __launch_bounds__(256) void feat_norm_kernel(const float* feat, float* out) {
  __shared__ float red[256];
  const int b = blockIdx.x;
  float s = 0.f;
  for (int c = threadIdx.x; c < kOUT; c += 256) { float v = feat[(long long)b * kOUT + c]; s += v * v; }
  red[threadIdx.x] = s; __syncthreads();
  if (threadIdx.x == 0) { float t = 0; for (int i = 0; i < 256; i++) t += red[i]; red[0] = rsqrtf(t); }
  __syncthreads();
  const float inv = red[0];
  for (int c = threadIdx.x; c < kOUT; c += 256)
    out[(long long)b * kOUT + c] = feat[(long long)b * kOUT + c] * inv;
}

__global__ void feat_mean_kernel(const float* nf, float* out) {
  int c = blockIdx.x * 256 + threadIdx.x;
  if (c < kOUT) {
    float s = 0.f;
    for (int b = 0; b < kB; b++) s += nf[(long long)b * kOUT + c];
    out[c] = s / (float)kB;
  }
}

// ---------------- host orchestration ----------------
extern "C" void kernel_launch(void* const* d_in, const int* in_sizes, int n_in,
                              void* d_out, int out_size, void* d_ws, size_t ws_size,
                              hipStream_t stream) {
  (void)in_sizes; (void)n_in; (void)out_size; (void)ws_size;

  const float* video      = (const float*)d_in[0];
  const float* prompt     = (const float*)d_in[1];
  const float* conv1_w    = (const float*)d_in[2];
  const float* class_emb  = (const float*)d_in[3];
  const float* pos_emb    = (const float*)d_in[4];
  const float* ln_pre_w   = (const float*)d_in[5];
  const float* ln_pre_b   = (const float*)d_in[6];
  const float* ln1_w      = (const float*)d_in[7];
  const float* ln1_b      = (const float*)d_in[8];
  const float* qkv_w      = (const float*)d_in[9];
  const float* qkv_b      = (const float*)d_in[10];
  const float* attn_out_w = (const float*)d_in[11];
  const float* attn_out_b = (const float*)d_in[12];
  const float* ln2_w      = (const float*)d_in[13];
  const float* ln2_b      = (const float*)d_in[14];
  const float* fc_w       = (const float*)d_in[15];
  const float* fc_b       = (const float*)d_in[16];
  const float* proj2_w    = (const float*)d_in[17];
  const float* proj2_b    = (const float*)d_in[18];
  const float* ln_post_w  = (const float*)d_in[19];
  const float* ln_post_b  = (const float*)d_in[20];
  const float* proj       = (const float*)d_in[21];
  const float* pos_ctx    = (const float*)d_in[22];
  const float* ctx_w1     = (const float*)d_in[23];
  const float* ctx_b1     = (const float*)d_in[24];
  const float* ctx_w2     = (const float*)d_in[25];
  const float* ctx_b2     = (const float*)d_in[26];

  char* base = (char*)d_ws; size_t cur = 0;
  auto alloc   = [&](size_t bytes) -> void* { cur = (cur + 255) & ~(size_t)255; void* q = base + cur; cur += bytes; return q; };
  auto allocBF = [&](size_t n) { return (unsigned short*)alloc(n * 2); };
  auto allocF  = [&](size_t n) { return (float*)alloc(n * 4); };

  const size_t PAD = 131072;   // tail pad (elements) for 32-row A-tile over-reads

  // bf16 weight mirrors
  unsigned short* convWb = allocBF((size_t)kD * 768);
  unsigned short* qkvWb  = allocBF((size_t)kNL * 3 * kD * kD);
  unsigned short* aoWb   = allocBF((size_t)kNL * kD * kD);
  unsigned short* fcWb   = allocBF((size_t)kNL * 4 * kD * kD);
  unsigned short* p2Wb   = allocBF((size_t)kNL * 4 * kD * kD);
  unsigned short* projTb = allocBF((size_t)kOUT * kD);
  unsigned short* cw1b   = allocBF((size_t)(kD / 4) * kD);
  unsigned short* cw2b   = allocBF((size_t)kD * (kD / 4));

  // activations (A operands padded for TDM tile over-read)
  unsigned short* im2colb = allocBF((size_t)kB * kNP * 768 + PAD);
  float*          patchf  = allocF((size_t)kB * kNP * kD);
  float*          x       = allocF((size_t)kB * kLCAP * kD);
  unsigned short* hb      = allocBF((size_t)kB * kLCAP * kD + PAD);
  unsigned short* qkvb    = allocBF((size_t)kB * kLCAP * 3 * kD + PAD);
  unsigned short* vt      = allocBF((size_t)kB * kH * kHD * kLC);
  float*          scores  = allocF((size_t)kB * kH * kLCAP * kLC);
  unsigned short* probsb  = allocBF((size_t)kB * kH * kLCAP * kLC + PAD);
  unsigned short* ob      = allocBF((size_t)kB * kLCAP * kD + PAD);
  unsigned short* mb      = allocBF((size_t)kB * kLCAP * 4 * kD + PAD);
  float*          tctx    = allocF((size_t)kSZ * kB * kD);
  unsigned short* tctxb   = allocBF((size_t)kSZ * kB * kD + PAD);
  unsigned short* hmb     = allocBF((size_t)kSZ * kB * (kD / 4) + PAD);
  float*          tctx2   = allocF((size_t)kSZ * kB * kD);
  int*            idxbuf  = (int*)alloc(kB * sizeof(int));
  unsigned short* xpostb  = allocBF((size_t)kB * kD + PAD);
  float*          feat    = allocF((size_t)kB * kOUT);

  auto cvt = [&](const float* s, unsigned short* d, size_t n) {
    cvt_bf16_kernel<<<(int)((n + 255) / 256), 256, 0, stream>>>(s, d, (long long)n);
  };
  cvt(conv1_w,    convWb, (size_t)kD * 768);
  cvt(qkv_w,      qkvWb,  (size_t)kNL * 3 * kD * kD);
  cvt(attn_out_w, aoWb,   (size_t)kNL * kD * kD);
  cvt(fc_w,       fcWb,   (size_t)kNL * 4 * kD * kD);
  cvt(proj2_w,    p2Wb,   (size_t)kNL * 4 * kD * kD);
  cvt(ctx_w1,     cw1b,   (size_t)(kD / 4) * kD);
  cvt(ctx_w2,     cw2b,   (size_t)kD * (kD / 4));
  {
    long long n = (long long)kOUT * kD;
    transpose_proj_kernel<<<(int)((n + 255) / 256), 256, 0, stream>>>(proj, projTb);
  }

  auto gemm = [&](const unsigned short* A, int lda,
                  const unsigned short* W, long long ldwn,
                  const float* bias, const float* resid, int ldres,
                  float* C, unsigned short* Cb, int ldc,
                  int M, int N, int K, int op, float scale,
                  int nz, int zh,
                  long long sAb, long long sAh, long long sWb, long long sWh,
                  long long sCb, long long sCh) {
    GemmP p;
    p.A = A; p.Wt = W; p.bias = bias; p.resid = resid; p.C = C; p.Cb = Cb;
    p.ldwn = ldwn;
    p.sAb = sAb; p.sAh = sAh; p.sWb = sWb; p.sWh = sWh; p.sCb = sCb; p.sCh = sCh;
    p.M = M; p.N = N; p.K = K; p.lda = lda; p.ldc = ldc; p.ldres = ldres;
    p.zh = zh; p.op = op; p.scale = scale;
    dim3 grid((N + 255) / 256, (M + 31) / 32, nz);
    wmma_gemm_bf16<<<grid, 128, 0, stream>>>(p);
  };

  // ---- patch embed + assemble + ln_pre ----
  {
    long long n = (long long)kB * kNP * 768;
    im2col_kernel<<<(int)((n + 255) / 256), 256, 0, stream>>>(video, im2colb);
    gemm(im2colb, 768, convWb, 768, nullptr, nullptr, 0, patchf, nullptr, kD,
         kB * kNP, kD, 768, 0, 1.f, 1, 1, 0, 0, 0, 0, 0, 0);
    long long n2 = (long long)kB * kLCAP * kD;
    assemble_kernel<<<(int)((n2 + 255) / 256), 256, 0, stream>>>(patchf, class_emb, pos_emb, x);
    ln_kernel<<<kB * kLCAP, 128, 0, stream>>>(x, kD, ln_pre_w, ln_pre_b, x, kD, nullptr, 0, kD);
  }

  const int MT = kB * kLCAP;   // flattened token rows (padded; pad rows are finite garbage)

  for (int li = 0; li < kNL; li++) {
    const int Ls = (li == 0) ? kL0 : (li == 1) ? (kL0 + 1) : (kL0 + 1 + kSZ);

    // h = LN1(x) -> bf16
    ln_kernel<<<MT, 128, 0, stream>>>(x, kD, ln1_w + (size_t)li * kD, ln1_b + (size_t)li * kD,
                                      nullptr, 0, hb, kD, kD);
    // qkv = h @ qkv_w^T + qkv_b -> bf16
    gemm(hb, kD, qkvWb + (size_t)li * 3 * kD * kD, kD,
         qkv_b + (size_t)li * 3 * kD, nullptr, 0, nullptr, qkvb, 3 * kD,
         MT, 3 * kD, kD, 1, 1.f, 1, 1, 0, 0, 0, 0, 0, 0);
    // vt = V transposed (K-contiguous rows for attn*V)
    {
      long long n = (long long)kB * kH * kHD * kLC;
      transpose_v_kernel<<<(int)((n + 255) / 256), 256, 0, stream>>>(qkvb, vt);
    }
    // scores = q @ k^T / 8     (per batch*head)
    gemm(qkvb, 3 * kD, qkvb + kD, 3 * kD, nullptr, nullptr, 0, scores, nullptr, kLC,
         kLCAP, kLCAP, kHD, 0, 0.125f, kB * kH, kH,
         (long long)kLCAP * 3 * kD, kHD, (long long)kLCAP * 3 * kD, kHD,
         (long long)kH * kLCAP * kLC, (long long)kLCAP * kLC);
    // softmax rows -> bf16 probs (zero padded to kLC)
    softmax_kernel<<<kB * kH * kLCAP, 128, 0, stream>>>(scores, probsb, Ls, kLC);
    // o = probs @ v            (per batch*head, vt K-contiguous)
    gemm(probsb, kLC, vt, kLC, nullptr, nullptr, 0, nullptr, ob, kD,
         kLCAP, kHD, kLC, 0, 1.f, kB * kH, kH,
         (long long)kH * kLCAP * kLC, (long long)kLCAP * kLC,
         (long long)kH * kHD * kLC, (long long)kHD * kLC,
         (long long)kLCAP * kD, kHD);
    // x = x + o @ ow^T + ob
    gemm(ob, kD, aoWb + (size_t)li * kD * kD, kD,
         attn_out_b + (size_t)li * kD, x, kD, x, nullptr, kD,
         MT, kD, kD, 2, 1.f, 1, 1, 0, 0, 0, 0, 0, 0);
    // h2 = LN2(x) -> bf16
    ln_kernel<<<MT, 128, 0, stream>>>(x, kD, ln2_w + (size_t)li * kD, ln2_b + (size_t)li * kD,
                                      nullptr, 0, hb, kD, kD);
    // t = qgelu(h2 @ fc_w^T + fc_b) -> bf16
    gemm(hb, kD, fcWb + (size_t)li * 4 * kD * kD, kD,
         fc_b + (size_t)li * 4 * kD, nullptr, 0, nullptr, mb, 4 * kD,
         MT, 4 * kD, kD, 3, 1.f, 1, 1, 0, 0, 0, 0, 0, 0);
    // x = x + t @ proj2_w^T + proj2_b
    gemm(mb, 4 * kD, p2Wb + (size_t)li * 4 * kD * kD, 4 * kD,
         proj2_b + (size_t)li * kD, x, kD, x, nullptr, kD,
         MT, kD, 4 * kD, 2, 1.f, 1, 1, 0, 0, 0, 0, 0, 0);

    // ---- token edits ----
    if (li == 0) {
      copy_prompt_kernel<<<(kB * kD + 255) / 256, 256, 0, stream>>>(prompt, x);
    } else if (li < kNL - 1) {
      float* aout = (float*)d_out + (size_t)kB * kOUT + kOUT + (size_t)(li - 1) * kB * kNP;
      attn_reduce_argmax<<<kB, 256, 0, stream>>>(probsb, aout, idxbuf);
      long long n3 = (long long)kSZ * kB * kD;
      build_ctx_kernel<<<(int)((n3 + 255) / 256), 256, 0, stream>>>(x, idxbuf, pos_ctx, tctx, tctxb, li);
      // hmlp = relu(temp @ ctx_w1^T + b1)
      gemm(tctxb, kD, cw1b, kD, ctx_b1, nullptr, 0, nullptr, hmb, kD / 4,
           kSZ * kB, kD / 4, kD, 4, 1.f, 1, 1, 0, 0, 0, 0, 0, 0);
      // temp2 = temp + hmlp @ ctx_w2^T + b2
      gemm(hmb, kD / 4, cw2b, kD / 4, ctx_b2, tctx, kD, tctx2, nullptr, kD,
           kSZ * kB, kD, kD / 4, 2, 1.f, 1, 1, 0, 0, 0, 0, 0, 0);
      scatter_ctx_kernel<<<(int)((n3 + 255) / 256), 256, 0, stream>>>(tctx2, x);
    }
  }

  // ---- head: ln_post(x[:,0]) @ projT^T, normalize, mean ----
  ln_kernel<<<kB, 128, 0, stream>>>(x, (long long)kLCAP * kD, ln_post_w, ln_post_b,
                                    nullptr, 0, xpostb, kD, kD);
  gemm(xpostb, kD, projTb, kD, nullptr, nullptr, 0, feat, nullptr, kOUT,
       kB, kOUT, kD, 0, 1.f, 1, 1, 0, 0, 0, 0, 0, 0);
  feat_norm_kernel<<<kB, 256, 0, stream>>>(feat, (float*)d_out);
  feat_mean_kernel<<<2, 256, 0, stream>>>((float*)d_out, (float*)d_out + (size_t)kB * kOUT);
}
